// GSSA_472446402917
// MI455X (gfx1250) — matrix operand order repaired and checked by reading, not compiled, not run on hardware
//
#include <hip/hip_runtime.h>

typedef __attribute__((ext_vector_type(16))) _Float16 v16h;
typedef __attribute__((ext_vector_type(8)))  float    v8f;

#define DEV static __device__ __forceinline__

// ---------- constants ----------
#define Bn     16
#define NGRP   1024          // 32*32 groups per image
#define SCALE  0.17677669529663687f   // 32^-0.5
#define LN_EPS 1e-5f

#define XSTR   264           // xs panel stride (halves): 528B = odd*16B, bank-safe
#define PSTR   40            // kbuf/vbuf/pbuf row stride (halves): 80B, bank-safe

// ---------- WMMA fragment helpers (gfx1250 wave32 layouts) ----------

// A fragment, 16x32 f16. element (m,k) at src[m*ld + k] (k contiguous)
DEV v16h load_a(const _Float16* src, int ld) {
  const int lane = threadIdx.x & 31;
  const int m  = lane & 15;
  const int kb = (lane < 16) ? 0 : 8;
  v16h a;
#pragma unroll
  for (int i = 0; i < 16; ++i) {
    const int v = i >> 1, p = i & 1;
    const int k = (v < 4) ? (kb + 2 * v + p) : (16 + kb + 2 * (v - 4) + p);
    a[i] = src[m * ld + k];
  }
  return a;
}

// B fragment, 32x16 f16 from N-major source: element (k,n) at src[n*ldn + k]
// -> each lane reads two runs of 8 contiguous halves (k = kb..kb+7, 16+kb..16+kb+7)
DEV v16h load_bt(const _Float16* src, int ldn) {
  const int lane = threadIdx.x & 31;
  const int n  = lane & 15;
  const int kb = (lane < 16) ? 0 : 8;
  v16h b;
#pragma unroll
  for (int i = 0; i < 16; ++i) {
    const int v = i >> 1, p = i & 1;
    const int k = (v < 4) ? (kb + 2 * v + p) : (16 + kb + 2 * (v - 4) + p);
    b[i] = src[n * ldn + k];
  }
  return b;
}

DEV v8f wmma_f16(v16h a, v16h b, v8f c) {
  return __builtin_amdgcn_wmma_f32_16x16x32_f16(false, a, false, b,
                                                (short)0, c, false, false);
}

// ---------- k0: f32 -> f16 weight conversion ----------
__global__ void gssa_k0_cvt(const float* __restrict__ src,
                            _Float16* __restrict__ dst, int n) {
  const int i = blockIdx.x * 256 + threadIdx.x;
  if (i < n) dst[i] = (_Float16)src[i];
}

// ---------- k1: per-group fused qkv GEMM + local attention + LN/GELU ----------
// grid = 16384 blocks (one per group), 256 threads (8 waves)
__global__ void gssa_k1_local(const float* __restrict__ x,
                              const float* __restrict__ gtok,
                              const float* __restrict__ lng,
                              const float* __restrict__ lnb,
                              const _Float16* __restrict__ wqkv_h,
                              _Float16* __restrict__ t_hT,
                              _Float16* __restrict__ gf_h) {
  __shared__ _Float16 xs[16 * XSTR];  // B operand, N-major: (n, c); n=0 token
  __shared__ float qkvb[768 * 10];    // qkv result
  __shared__ float dloc[8 * 120];     // per-head 10x10 dots (stride 12)
  __shared__ float obuf[8 * 320];     // per-head 10x32 attention output

  const int gid = blockIdx.x;
  const int b  = gid >> 10, ng = gid & 1023;
  const int xi = ng >> 5,  yi = ng & 31;
  const int tid = threadIdx.x;

  // stage group tile (f16), N-major; pad cols 10..15 with zeros
  for (int e = tid; e < 256 * 16; e += 256) {
    const int c = e >> 4, n = e & 15;
    float v = 0.f;
    if (n == 0) v = gtok[c];
    else if (n <= 9) {
      const int p  = n - 1;
      const int hr = xi * 3 + p / 3, wc = yi * 3 + p % 3;
      v = x[((size_t)(b * 256 + c) * 96 + hr) * 96 + wc];
    }
    xs[n * XSTR + c] = (_Float16)v;
  }
  __syncthreads();

  const int wv   = tid >> 5, lane = tid & 31;
  const int hf   = lane >> 4, col = lane & 15;

  // qkv GEMM: 48 M-tiles of 16 rows, 8 waves x 6 tiles, K=256 in 8 chunks
#pragma unroll
  for (int t = 0; t < 6; ++t) {
    const int mt = wv + (t << 3);
    v8f acc = {};
#pragma unroll
    for (int kk = 0; kk < 8; ++kk) {
      const v16h a  = load_a(wqkv_h + (size_t)(mt * 16) * 256 + kk * 32, 256);
      const v16h bb = load_bt(xs + kk * 32, XSTR);
      acc = wmma_f16(a, bb, acc);
    }
    if (col < 10) {
#pragma unroll
      for (int r = 0; r < 8; ++r)
        qkvb[(mt * 16 + r + hf * 8) * 10 + col] = acc[r];
    }
  }
  __syncthreads();

  // per-head 10x10 attention; head = wave id
  const int head = wv;
  const int qo = head * 32, ko = 256 + head * 32, vo = 512 + head * 32;
  for (int e = lane; e < 100; e += 32) {
    const int i = e / 10, j = e % 10;
    float s = 0.f;
    for (int dd = 0; dd < 32; ++dd)
      s += qkvb[(qo + dd) * 10 + i] * qkvb[(ko + dd) * 10 + j];
    dloc[head * 120 + i * 12 + j] = s * SCALE;
  }
  __syncthreads();
  if (lane < 10) {
    float mx = -1e30f;
    for (int j = 0; j < 10; ++j) mx = fmaxf(mx, dloc[head * 120 + lane * 12 + j]);
    float ev[10], sm = 0.f;
    for (int j = 0; j < 10; ++j) {
      ev[j] = __expf(dloc[head * 120 + lane * 12 + j] - mx);
      sm += ev[j];
    }
    const float inv = 1.f / sm;
    for (int j = 0; j < 10; ++j) dloc[head * 120 + lane * 12 + j] = ev[j] * inv;
  }
  __syncthreads();
  for (int e = lane; e < 320; e += 32) {
    const int i = e >> 5, dd = e & 31;
    float s = 0.f;
    for (int j = 0; j < 10; ++j)
      s += dloc[head * 120 + i * 12 + j] * qkvb[(vo + dd) * 10 + j];
    obuf[head * 320 + e] = s;
  }
  __syncthreads();

  // group token: LayerNorm over d (32 lanes) + exact GELU
  // -> t stored (b, n, c) so k2's B fragments are k-contiguous
  {
    const float v0 = obuf[head * 320 + lane];
    float s = v0;
#pragma unroll
    for (int m = 16; m >= 1; m >>= 1) s += __shfl_xor(s, m, 32);
    const float mu = s * (1.f / 32.f);
    const float dv = v0 - mu;
    float s2 = dv * dv;
#pragma unroll
    for (int m = 16; m >= 1; m >>= 1) s2 += __shfl_xor(s2, m, 32);
    const float var = s2 * (1.f / 32.f);
    float gn = dv * rsqrtf(var + LN_EPS) * lng[lane] + lnb[lane];
    gn = 0.5f * gn * (1.f + erff(gn * 0.70710678118654752f));
    t_hT[((size_t)(b * 1024 + ng)) * 256 + head * 32 + lane] = (_Float16)gn;
  }
  // window features -> gf (b,h, j=ng, col=p*32+dd) (coalesced store)
  for (int e = lane; e < 288; e += 32) {
    const int p = e >> 5, dd = e & 31;
    gf_h[((size_t)((b * 8 + head) * 1024 + ng)) * 288 + e] =
        (_Float16)obuf[head * 320 + (p + 1) * 32 + dd];
  }
}

// ---------- k2: qk GEMM (512x256 x 1024 per image) ----------
// grid = 4096 blocks x 256 threads; one 16x16 tile per wave
__global__ void gssa_k2_qk(const _Float16* __restrict__ wqk_h,
                           const float* __restrict__ bqk,
                           const _Float16* __restrict__ t_hT,
                           _Float16* __restrict__ wq_h,
                           _Float16* __restrict__ wk_h) {
  const int wid  = blockIdx.x * 8 + (threadIdx.x >> 5);
  const int lane = threadIdx.x & 31, hf = lane >> 4, col = lane & 15;
  const int b   = wid >> 11;        // / (32*64)
  const int rem = wid & 2047;
  const int mt  = rem >> 6, nt = rem & 63;
  v8f acc = {};
#pragma unroll
  for (int kk = 0; kk < 8; ++kk) {
    const v16h a  = load_a(wqk_h + (size_t)(mt * 16) * 256 + kk * 32, 256);
    const v16h bb = load_bt(t_hT + ((size_t)(b * 1024 + nt * 16)) * 256 + kk * 32, 256);
    acc = wmma_f16(a, bb, acc);
  }
  const int n = nt * 16 + col;
#pragma unroll
  for (int r = 0; r < 8; ++r) {
    const int m = mt * 16 + r + hf * 8;
    const int head = m >> 6, rr = m & 63;
    const float v = acc[r] + bqk[m];
    if (rr < 32)
      wq_h[((size_t)((b * 8 + head) * 1024 + n)) * 32 + rr] = (_Float16)(v * SCALE);
    else
      wk_h[((size_t)((b * 8 + head) * 1024 + n)) * 32 + (rr - 32)] = (_Float16)v;
  }
}

// ---------- k3: flash window attention + agg + scatter to fmap ----------
// grid = 1024 blocks (128 (b,h) x 8 i-blocks) x 256 threads; wave = 16-row i-tile
__global__ void gssa_k3_flash(const _Float16* __restrict__ wq_h,
                              const _Float16* __restrict__ wk_h,
                              const _Float16* __restrict__ gf_h,
                              _Float16* __restrict__ fmapT) {
  __shared__ _Float16 kbuf[32 * PSTR];       // K chunk: (jl, dd), padded rows
  __shared__ _Float16 vbuf[288 * PSTR];      // V chunk, N-major: (col, jl)
  __shared__ _Float16 pbuf[8 * 16 * PSTR];   // P staging per wave, padded rows

  const int bh = blockIdx.x >> 3, iblk = blockIdx.x & 7;
  const int b  = bh >> 3, head = bh & 7;
  const int tid = threadIdx.x, wv = tid >> 5, lane = tid & 31;
  const int hf = lane >> 4, col = lane & 15;
  const int i0 = iblk * 128 + wv * 16;
  const size_t bhNG = (size_t)bh * 1024;

  const v16h aq = load_a(wq_h + (bhNG + i0) * 32, 32);

  v8f acc[18];
#pragma unroll
  for (int t = 0; t < 18; ++t) acc[t] = (v8f){};
  float mr[8], lr[8];
#pragma unroll
  for (int r = 0; r < 8; ++r) { mr[r] = -1e30f; lr[r] = 0.f; }

  for (int jc = 0; jc < 32; ++jc) {
    const int j0 = jc * 32;
    __syncthreads();   // previous chunk fully consumed
    for (int e = tid; e < 1024; e += 256)
      kbuf[(e >> 5) * PSTR + (e & 31)] =
          wk_h[(bhNG + j0 + (e >> 5)) * 32 + (e & 31)];
    for (int e = tid; e < 32 * 288; e += 256) {
      const int jl = e / 288, cl = e % 288;   // global reads coalesced over cl
      vbuf[cl * PSTR + jl] = gf_h[(bhNG + j0 + jl) * 288 + cl];
    }
    // prefetch next chunk's V and K panels into cache (global_prefetch_b8)
    if (jc < 31) {
      const char* vnext = (const char*)(gf_h + (bhNG + j0 + 32) * 288);
      if (tid < 144) __builtin_prefetch(vnext + tid * 128, 0, 1);
      const char* knext = (const char*)(wk_h + (bhNG + j0 + 32) * 32);
      if (tid < 16) __builtin_prefetch(knext + tid * 128, 0, 1);
    }
    __syncthreads();

    const v16h bk0 = load_bt(kbuf, PSTR);
    const v16h bk1 = load_bt(kbuf + 16 * PSTR, PSTR);
    const v8f zz = {};
    const v8f s0 = wmma_f16(aq, bk0, zz);
    const v8f s1 = wmma_f16(aq, bk1, zz);

    float sf[8];
#pragma unroll
    for (int r = 0; r < 8; ++r) {
      float mx = fmaxf(s0[r], s1[r]);
#pragma unroll
      for (int m = 8; m >= 1; m >>= 1) mx = fmaxf(mx, __shfl_xor(mx, m, 32));
      const float mn = fmaxf(mr[r], mx);
      sf[r] = __expf(mr[r] - mn);
      const float p0 = __expf(s0[r] - mn);
      const float p1 = __expf(s1[r] - mn);
      float rs = p0 + p1;
#pragma unroll
      for (int m = 8; m >= 1; m >>= 1) rs += __shfl_xor(rs, m, 32);
      lr[r] = lr[r] * sf[r] + rs;
      mr[r] = mn;
      pbuf[wv * 16 * PSTR + (r + hf * 8) * PSTR + col]      = (_Float16)p0;
      pbuf[wv * 16 * PSTR + (r + hf * 8) * PSTR + 16 + col] = (_Float16)p1;
    }
#pragma unroll
    for (int t = 0; t < 18; ++t)
#pragma unroll
      for (int r = 0; r < 8; ++r) acc[t][r] = acc[t][r] * sf[r];

    __syncthreads();   // pbuf visible in A layout
    const v16h ap = load_a(pbuf + wv * 16 * PSTR, PSTR);
#pragma unroll
    for (int t = 0; t < 18; ++t) {
      const v16h bv = load_bt(vbuf + t * 16 * PSTR, PSTR);
      acc[t] = wmma_f16(ap, bv, acc[t]);
    }
  }

  float inv[8];
#pragma unroll
  for (int r = 0; r < 8; ++r) inv[r] = 1.f / lr[r];
  // store fmap as (b, hw, c): 16-lane-contiguous writes, k-contiguous for k4
#pragma unroll
  for (int t = 0; t < 18; ++t) {
#pragma unroll
    for (int r = 0; r < 8; ++r) {
      const int i  = i0 + r + hf * 8;         // group index
      const int c  = t * 16 + col;            // p*32 + dd
      const int p  = c >> 5, dd = c & 31;
      const int xi = i >> 5, yi = i & 31;
      const int hr = xi * 3 + p / 3, wc = yi * 3 + p % 3;
      fmapT[((size_t)b * 9216 + hr * 96 + wc) * 256 + head * 32 + dd] =
          (_Float16)(acc[t][r] * inv[r]);
    }
  }
}

// ---------- k4: output GEMM (256x256 x 9216 per image) + bias ----------
// grid = 18432 blocks x 256 threads; one 16x16 tile per wave
__global__ void gssa_k4_out(const _Float16* __restrict__ wout_h,
                            const float* __restrict__ bout,
                            const _Float16* __restrict__ fmapT,
                            float* __restrict__ out) {
  const int wid  = blockIdx.x * 8 + (threadIdx.x >> 5);
  const int lane = threadIdx.x & 31, hf = lane >> 4, col = lane & 15;
  const int b   = wid / (16 * 576);
  const int rem = wid % (16 * 576);
  const int mt  = rem / 576, nt = rem % 576;
  v8f acc = {};
#pragma unroll
  for (int kk = 0; kk < 8; ++kk) {
    const v16h a  = load_a(wout_h + (size_t)(mt * 16) * 256 + kk * 32, 256);
    const v16h bb = load_bt(fmapT + ((size_t)b * 9216 + nt * 16) * 256 + kk * 32, 256);
    acc = wmma_f16(a, bb, acc);
  }
  const int n = nt * 16 + col;
#pragma unroll
  for (int r = 0; r < 8; ++r) {
    const int m = mt * 16 + r + hf * 8;
    out[((size_t)(b * 256 + m)) * 9216 + n] = acc[r] + bout[m];
  }
}

// ---------- host launcher ----------
extern "C" void kernel_launch(void* const* d_in, const int* in_sizes, int n_in,
                              void* d_out, int out_size, void* d_ws, size_t ws_size,
                              hipStream_t stream) {
  (void)in_sizes; (void)n_in; (void)out_size; (void)ws_size;
  const float* x    = (const float*)d_in[0];
  const float* wqkv = (const float*)d_in[1];
  const float* gtok = (const float*)d_in[2];
  const float* lng  = (const float*)d_in[3];
  const float* lnb  = (const float*)d_in[4];
  const float* wqk  = (const float*)d_in[5];
  const float* bqk  = (const float*)d_in[6];
  const float* wout = (const float*)d_in[7];
  const float* bout = (const float*)d_in[8];
  float* out = (float*)d_out;

  char* ws = (char*)d_ws;
  _Float16* wqkv_h = (_Float16*)(ws);                 // 768*256      = 393216 B
  _Float16* wqk_h  = (_Float16*)(ws + 393216);        // 512*256      = 262144 B
  _Float16* wout_h = (_Float16*)(ws + 655360);        // 256*256      = 131072 B
  _Float16* t_hT   = (_Float16*)(ws + 786432);        // (B, NG, 256) = 8 MB
  _Float16* wq_h   = (_Float16*)(ws + 9175040);       // (B,h,NG,32)  = 8 MB
  _Float16* wk_h   = (_Float16*)(ws + 17563648);      // 8 MB
  _Float16* gf_h   = (_Float16*)(ws + 25952256);      // (B,h,NG,288) = 75.5 MB
  _Float16* fmapT  = (_Float16*)(ws + 101449728);     // (B, HW, 256) = 75.5 MB

  gssa_k0_cvt<<<768, 256, 0, stream>>>(wqkv, wqkv_h, 768 * 256);
  gssa_k0_cvt<<<512, 256, 0, stream>>>(wqk,  wqk_h,  512 * 256);
  gssa_k0_cvt<<<256, 256, 0, stream>>>(wout, wout_h, 256 * 256);

  gssa_k1_local<<<Bn * NGRP, 256, 0, stream>>>(x, gtok, lng, lnb, wqkv_h, t_hT, gf_h);
  gssa_k2_qk<<<4096, 256, 0, stream>>>(wqk_h, bqk, t_hT, wq_h, wk_h);
  gssa_k3_flash<<<1024, 256, 0, stream>>>(wq_h, wk_h, gf_h, fmapT);
  gssa_k4_out<<<18432, 256, 0, stream>>>(wout_h, bout, fmapT, out);
}
